// DiffLlamaFlashAttention2_64544768524529
// MI455X (gfx1250) — matrix operand order repaired and checked by reading, not compiled, not run on hardware
//
#include <hip/hip_runtime.h>
#include <hip/hip_bf16.h>

// ---------------- problem constants ----------------
constexpr int S_   = 2048;   // sequence length
constexpr int HID  = 2048;   // hidden size
constexpr int H_   = 32;     // query heads
constexpr int HKV  = 8;      // kv heads
constexpr int D_   = 64;     // head dim
constexpr int DV   = 128;    // concatenated v dim per pair
constexpr float NEGC        = -1.0e9f;
constexpr float LAMBDA_INIT = 0.74556922802635f;     // 0.8 - 0.6*exp(-2.4)
constexpr float ONE_M_LI    = 0.25443077197365f;     // 1 - LAMBDA_INIT
constexpr float SCALE_QK    = 0.125f;                // 1/sqrt(64)

typedef __attribute__((ext_vector_type(16))) __bf16 v16bf;
typedef __attribute__((ext_vector_type(8)))  __bf16 v8bf;
typedef __attribute__((ext_vector_type(8)))  float  v8f;

// D = A(16x32) * B(32x16) + C, bf16 in / f32 accum
__device__ __forceinline__ v8f wmma_bf16(v16bf a, v16bf b, v8f c) {
  return __builtin_amdgcn_wmma_f32_16x16x32_bf16(false, a, false, b,
                                                 (short)0, c, false, false);
}

// A-fragment (16x32, 16-bit): lane = row (l&15), half h=l>>4 holds
// K runs [8h, 8h+8) and [16+8h, 16+8h+8)   (ISA 7.12.2)
__device__ __forceinline__ v16bf ldfragA(const __bf16* base, int ld, int lane) {
  int r = lane & 15, h = lane >> 4;
  const __bf16* s = base + (size_t)r * ld + 8 * h;
  union { v16bf v; v8bf p[2]; } u;
  u.p[0] = *(const v8bf*)(s);
  u.p[1] = *(const v8bf*)(s + 16);
  return u.v;
}

// B-fragment (32x16, 16-bit): lane = column (l&15), half h=l>>4 holds
// K runs [8h, 8h+8) and [16+8h, 16+8h+8)   (same order as A)
__device__ __forceinline__ v16bf ldfragB(const __bf16* base, int ld, int lane) {
  int r = lane & 15, h = lane >> 4;
  const __bf16* s = base + (size_t)r * ld + 8 * h;
  union { v16bf v; v8bf p[2]; } u;
  u.p[0] = *(const v8bf*)(s);
  u.p[1] = *(const v8bf*)(s + 16);
  return u.v;
}

// ---------------- fp32 -> bf16 convert ----------------
__global__ __launch_bounds__(256) void cvt_bf16_k(const float* __restrict__ in,
                                                  __bf16* __restrict__ out, int n) {
  int i = blockIdx.x * 256 + threadIdx.x;
  if (i < n) out[i] = (__bf16)in[i];
}

// ---------------- lambda scalar ----------------
__global__ __launch_bounds__(32) void lam_k(const float* q1, const float* k1,
                                            const float* q2, const float* k2,
                                            float* out) {
  int t = threadIdx.x;
  float s1 = q1[t] * k1[t] + q1[t + 32] * k1[t + 32];
  float s2 = q2[t] * k2[t] + q2[t + 32] * k2[t + 32];
  #pragma unroll
  for (int o = 1; o < 32; o <<= 1) {
    s1 += __shfl_xor(s1, o, 32);
    s2 += __shfl_xor(s2, o, 32);
  }
  if (t == 0) out[0] = __expf(s1) - __expf(s2) + LAMBDA_INIT;
}

// ---------------- generic bf16 WMMA GEMM:  C[m][n] = sum_k A[m][k]*W[n][k] ----
// one wave computes a 32x64 tile of C
__global__ __launch_bounds__(32) void gemm_bf16_nt(const __bf16* __restrict__ A,
                                                   const __bf16* __restrict__ W,
                                                   void* __restrict__ C,
                                                   int M, int N, int K, int f32out) {
  int lane = threadIdx.x;
  int n0 = blockIdx.x * 64, m0 = blockIdx.y * 32;
  v8f acc[2][4];
  #pragma unroll
  for (int mt = 0; mt < 2; ++mt)
    #pragma unroll
    for (int nt = 0; nt < 4; ++nt) acc[mt][nt] = (v8f)0.0f;

  for (int k0 = 0; k0 < K; k0 += 32) {
    __builtin_prefetch(A + (size_t)m0 * K + k0 + 128, 0, 1);   // global_prefetch_b8
    v16bf a0 = ldfragA(A + (size_t)m0 * K + k0, K, lane);
    v16bf a1 = ldfragA(A + (size_t)(m0 + 16) * K + k0, K, lane);
    #pragma unroll
    for (int nt = 0; nt < 4; ++nt) {
      v16bf b = ldfragB(W + (size_t)(n0 + 16 * nt) * K + k0, K, lane);
      acc[0][nt] = wmma_bf16(a0, b, acc[0][nt]);
      acc[1][nt] = wmma_bf16(a1, b, acc[1][nt]);
    }
  }
  int col = lane & 15, h = lane >> 4;
  #pragma unroll
  for (int mt = 0; mt < 2; ++mt)
    #pragma unroll
    for (int nt = 0; nt < 4; ++nt)
      #pragma unroll
      for (int r = 0; r < 8; ++r) {
        int row = m0 + 16 * mt + r + 8 * h;
        int cc  = n0 + 16 * nt + col;
        float v = acc[mt][nt][r];
        if (f32out) ((float*)C)[(size_t)row * N + cc] = v;
        else        ((__bf16*)C)[(size_t)row * N + cc] = (__bf16)v;
      }
}

// ---------------- RoPE + relayout ----------------
// Qtmp (S, H*D) -> Qb (H, S, D) rotated
__global__ __launch_bounds__(256) void rope_k(const __bf16* __restrict__ Xtmp,
                                              const float* __restrict__ cosp,
                                              const float* __restrict__ sinp,
                                              __bf16* __restrict__ Xout,
                                              int nheads, int n) {
  int i = blockIdx.x * 256 + threadIdx.x;
  if (i >= n) return;
  int h = i / (S_ * D_);
  int rem = i - h * (S_ * D_);
  int s = rem / D_, d = rem - s * D_;
  int ld = nheads * D_;
  float x   = (float)Xtmp[(size_t)s * ld + h * D_ + d];
  float rot = (d < 32) ? -(float)Xtmp[(size_t)s * ld + h * D_ + d + 32]
                       :  (float)Xtmp[(size_t)s * ld + h * D_ + d - 32];
  float v = x * cosp[s * D_ + d] + rot * sinp[s * D_ + d];
  Xout[(size_t)h * S_ * D_ + (size_t)s * D_ + d] = (__bf16)v;
}

// Vtmp (S, HKV*D) -> Vt (HKV, D, S)  (transposed so PV B-frags are contiguous)
__global__ __launch_bounds__(256) void vtrans_k(const __bf16* __restrict__ Vtmp,
                                                __bf16* __restrict__ Vt, int n) {
  int i = blockIdx.x * 256 + threadIdx.x;
  if (i >= n) return;
  int kv = i / (S_ * D_);
  int rem = i - kv * (S_ * D_);
  int s = rem / D_, d = rem - s * D_;
  Vt[((size_t)kv * D_ + d) * S_ + s] = Vtmp[(size_t)s * (HKV * D_) + kv * D_ + d];
}

// ---------------- flash attention core ----------------
// scores for one head over a 32-key block; online-softmax update; P -> LDS (bf16)
__device__ __forceinline__ void head_block(const v16bf* aq, const __bf16* Krow,
                                           float* m, float* l, v8f* acc,
                                           __bf16* ldsP, int q0, int kb, int lane) {
  int col = lane & 15, h = lane >> 4;
  v8f s[2];
  #pragma unroll
  for (int t = 0; t < 2; ++t) {
    v16bf b0 = ldfragB(Krow + (16 * t) * D_ +  0, D_, lane);
    v16bf b1 = ldfragB(Krow + (16 * t) * D_ + 32, D_, lane);
    v8f z = (v8f)0.0f;
    s[t] = wmma_bf16(aq[0], b0, z);
    s[t] = wmma_bf16(aq[1], b1, s[t]);
  }
  // scale + causal mask (C layout: lane = key col, VGPR r = q row r+8h)
  #pragma unroll
  for (int t = 0; t < 2; ++t)
    #pragma unroll
    for (int r = 0; r < 8; ++r) {
      int key = kb + 16 * t + col;
      int row = q0 + r + 8 * h;
      float v = s[t][r] * SCALE_QK;
      s[t][r] = (key > row) ? NEGC : v;
    }
  #pragma unroll
  for (int r = 0; r < 8; ++r) {
    float bm = fmaxf(s[0][r], s[1][r]);
    #pragma unroll
    for (int o = 1; o < 16; o <<= 1) bm = fmaxf(bm, __shfl_xor(bm, o, 32));
    float mn = fmaxf(m[r], bm);
    float al = __expf(m[r] - mn);
    m[r] = mn;
    float rs = 0.0f;
    #pragma unroll
    for (int t = 0; t < 2; ++t) {
      float p = __expf(s[t][r] - mn);
      s[t][r] = p;
      rs += p;
    }
    #pragma unroll
    for (int o = 1; o < 16; o <<= 1) rs += __shfl_xor(rs, o, 32);
    l[r] = l[r] * al + rs;
    #pragma unroll
    for (int tt = 0; tt < 8; ++tt) acc[tt][r] *= al;
    #pragma unroll
    for (int t = 0; t < 2; ++t)
      ldsP[(r + 8 * h) * 32 + 16 * t + col] = (__bf16)s[t][r];
  }
}

__global__ __launch_bounds__(32) void attn_k(const __bf16* __restrict__ Qb,
                                             const __bf16* __restrict__ Kb,
                                             const __bf16* __restrict__ Vt,
                                             const float* __restrict__ lamp,
                                             __bf16* __restrict__ Ab) {
  __shared__ __bf16 ldsP1[16 * 32];
  __shared__ __bf16 ldsP2[16 * 32];
  int lane = threadIdx.x;
  int col = lane & 15, h = lane >> 4;
  int qt = blockIdx.x, pair = blockIdx.y;
  int q0 = qt * 16;
  int kva = pair >> 2, kvb = 4 + (pair >> 2);

  const __bf16* Q1 = Qb + ((size_t)pair        * S_ + q0) * D_;
  const __bf16* Q2 = Qb + ((size_t)(pair + 16) * S_ + q0) * D_;
  v16bf aq1[2], aq2[2];
  aq1[0] = ldfragA(Q1,      D_, lane); aq1[1] = ldfragA(Q1 + 32, D_, lane);
  aq2[0] = ldfragA(Q2,      D_, lane); aq2[1] = ldfragA(Q2 + 32, D_, lane);
  float lam = lamp[0];

  float m1[8], l1[8], m2[8], l2[8];
  v8f acc1[8], acc2[8];
  #pragma unroll
  for (int r = 0; r < 8; ++r) { m1[r] = -3.0e38f; m2[r] = -3.0e38f; l1[r] = 0.f; l2[r] = 0.f; }
  #pragma unroll
  for (int t = 0; t < 8; ++t) { acc1[t] = (v8f)0.0f; acc2[t] = (v8f)0.0f; }

  for (int kb = 0; kb < q0 + 16; kb += 32) {
    head_block(aq1, Kb + ((size_t)kva * S_ + kb) * D_, m1, l1, acc1, ldsP1, q0, kb, lane);
    head_block(aq2, Kb + ((size_t)kvb * S_ + kb) * D_, m2, l2, acc2, ldsP2, q0, kb, lane);
    v16bf pa1 = ldfragA(ldsP1, 32, lane);
    v16bf pa2 = ldfragA(ldsP2, 32, lane);
    #pragma unroll
    for (int t = 0; t < 8; ++t) {
      const __bf16* vp = Vt + ((size_t)((t < 4 ? kva : kvb) * D_ + (t & 3) * 16)) * S_ + kb;
      v16bf vf = ldfragB(vp, S_, lane);
      acc1[t] = wmma_bf16(pa1, vf, acc1[t]);
      acc2[t] = wmma_bf16(pa2, vf, acc2[t]);
    }
  }

  // epilogue: out1 - lam*out2, RMS norm over 128, scale by (1-lambda_init)
  float inv1[8], inv2[8], ss[8];
  #pragma unroll
  for (int r = 0; r < 8; ++r) { inv1[r] = 1.0f / l1[r]; inv2[r] = lam / l2[r]; ss[r] = 0.f; }
  #pragma unroll
  for (int t = 0; t < 8; ++t)
    #pragma unroll
    for (int r = 0; r < 8; ++r) {
      float v = acc1[t][r] * inv1[r] - acc2[t][r] * inv2[r];
      acc1[t][r] = v;
      ss[r] += v * v;
    }
  #pragma unroll
  for (int r = 0; r < 8; ++r) {
    float s = ss[r];
    #pragma unroll
    for (int o = 1; o < 16; o <<= 1) s += __shfl_xor(s, o, 32);
    float scl = rsqrtf(s * (1.0f / 128.0f) + 1e-6f) * ONE_M_LI;
    int row = q0 + r + 8 * h;
    #pragma unroll
    for (int t = 0; t < 8; ++t) {
      int c = pair * 128 + 16 * t + col;
      Ab[(size_t)row * (H_ * D_) + c] = (__bf16)(acc1[t][r] * scl);
    }
  }
}

// ---------------- host launcher ----------------
extern "C" void kernel_launch(void* const* d_in, const int* in_sizes, int n_in,
                              void* d_out, int out_size, void* d_ws, size_t ws_size,
                              hipStream_t stream) {
  const float* hs   = (const float*)d_in[0];
  const float* cosp = (const float*)d_in[1];
  const float* sinp = (const float*)d_in[2];
  const float* Wq   = (const float*)d_in[3];
  const float* Wk   = (const float*)d_in[4];
  const float* Wv   = (const float*)d_in[5];
  const float* Wo   = (const float*)d_in[6];
  const float* lq1  = (const float*)d_in[7];
  const float* lk1  = (const float*)d_in[8];
  const float* lq2  = (const float*)d_in[9];
  const float* lk2  = (const float*)d_in[10];

  char* ws = (char*)d_ws;
  // workspace layout (bytes)
  __bf16* hsb  = (__bf16*)(ws + 0);          // 2048*2048*2 = 8 MiB
  __bf16* Wqb  = (__bf16*)(ws + (8u  << 20));
  __bf16* Wkb  = (__bf16*)(ws + (16u << 20));     // 2 MiB
  __bf16* Wvb  = (__bf16*)(ws + (18u << 20));     // 2 MiB
  __bf16* Wob  = (__bf16*)(ws + (20u << 20));     // 8 MiB
  __bf16* Qtmp = (__bf16*)(ws + (28u << 20));     // 8 MiB (reused as Ab)
  __bf16* Ktmp = (__bf16*)(ws + (36u << 20));     // 2 MiB
  __bf16* Vtmp = (__bf16*)(ws + (38u << 20));     // 2 MiB
  __bf16* Qb   = (__bf16*)(ws + (40u << 20));     // 8 MiB
  __bf16* Kb   = (__bf16*)(ws + (48u << 20));     // 2 MiB
  __bf16* Vt   = (__bf16*)(ws + (50u << 20));     // 2 MiB
  float*  lamp = (float*) (ws + (52u << 20));
  __bf16* Ab   = Qtmp;                            // overlay: Qtmp dead after RoPE

  auto blocks = [](int n) { return (n + 255) / 256; };
  const int nHS = S_ * HID, nWq = H_ * D_ * HID, nWkv = HKV * D_ * HID, nWo = HID * H_ * D_;

  cvt_bf16_k<<<blocks(nHS),  256, 0, stream>>>(hs, hsb, nHS);
  cvt_bf16_k<<<blocks(nWq),  256, 0, stream>>>(Wq, Wqb, nWq);
  cvt_bf16_k<<<blocks(nWkv), 256, 0, stream>>>(Wk, Wkb, nWkv);
  cvt_bf16_k<<<blocks(nWkv), 256, 0, stream>>>(Wv, Wvb, nWkv);
  cvt_bf16_k<<<blocks(nWo),  256, 0, stream>>>(Wo, Wob, nWo);

  lam_k<<<1, 32, 0, stream>>>(lq1, lk1, lq2, lk2, lamp);

  // projections: C = hs @ W^T
  gemm_bf16_nt<<<dim3(2048 / 64, 2048 / 32), 32, 0, stream>>>(hsb, Wqb, Qtmp, 2048, 2048, 2048, 0);
  gemm_bf16_nt<<<dim3( 512 / 64, 2048 / 32), 32, 0, stream>>>(hsb, Wkb, Ktmp, 2048,  512, 2048, 0);
  gemm_bf16_nt<<<dim3( 512 / 64, 2048 / 32), 32, 0, stream>>>(hsb, Wvb, Vtmp, 2048,  512, 2048, 0);

  rope_k<<<blocks(H_ * S_ * D_),  256, 0, stream>>>(Qtmp, cosp, sinp, Qb, H_,  H_ * S_ * D_);
  rope_k<<<blocks(HKV * S_ * D_), 256, 0, stream>>>(Ktmp, cosp, sinp, Kb, HKV, HKV * S_ * D_);
  vtrans_k<<<blocks(HKV * S_ * D_), 256, 0, stream>>>(Vtmp, Vt, HKV * S_ * D_);

  attn_k<<<dim3(S_ / 16, 16), 32, 0, stream>>>(Qb, Kb, Vt, lamp, Ab);

  // out = Ab @ Wo^T  (fp32 output)
  gemm_bf16_nt<<<dim3(2048 / 64, 2048 / 32), 32, 0, stream>>>(Ab, Wob, d_out, 2048, 2048, 2048, 1);
}
